// LSTMEncoder_13683765805749
// MI455X (gfx1250) — compile-verified, matrix operand order checked by reading, not run on hardware
//
#include <hip/hip_runtime.h>
#include <hip/hip_bf16.h>

// ---------------------------------------------------------------------------
// LSTM (CudnnCompatible, gate order i,j,f,o) for B=64, T=512, D=1024.
//   * One-time: convert x -> bf16, transpose+convert W -> Wt[4096][2048] bf16.
//   * Per timestep: one fused kernel: z = [x_t, h_{t-1}] @ W via
//     v_wmma_f32_16x16x32_bf16 with LDS-staged, double-buffered A panel
//     (software pipelined), gate exchange through LDS, nonlinearities +
//     length masking, c/h update (h double-buffered bf16 across steps).
//   * Wt (16 MB bf16) is L2-resident across all 512 steps (192 MB L2).
// ---------------------------------------------------------------------------

typedef __attribute__((ext_vector_type(16))) __bf16 v16bf;
typedef __attribute__((ext_vector_type(8)))  float  v8f;

union Frag16 { v16bf v; uint4 q[2]; };

#define BATCH 64
#define TSTEPS 512
#define DIM 1024
#define KDIM 2048      // 2*DIM
#define NDIM 4096      // 4*DIM
#define APAD 136       // 128 + 8 bf16 pad: rotates banks 4/row -> conflict-free

// ------------------------- one-time preparation ----------------------------

__global__ __launch_bounds__(256) void convert_x_kernel(const float4* __restrict__ x,
                                                        __bf16* __restrict__ xbf) {
    size_t i = (size_t)blockIdx.x * blockDim.x + threadIdx.x;   // over nelem/4
    float4 v = x[i];
    size_t o = i * 4;
    xbf[o + 0] = (__bf16)v.x;
    xbf[o + 1] = (__bf16)v.y;
    xbf[o + 2] = (__bf16)v.z;
    xbf[o + 3] = (__bf16)v.w;
}

// W is [K=2048, N=4096] row-major f32 -> Wt [N=4096, K=2048] bf16
__global__ __launch_bounds__(256) void transpose_w_kernel(const float* __restrict__ w,
                                                          __bf16* __restrict__ wt) {
    __shared__ float tile[32][33];
    const int kb = blockIdx.x * 32;
    const int nb = blockIdx.y * 32;
    const int tx = threadIdx.x & 31;
    const int ty = threadIdx.x >> 5;
    #pragma unroll
    for (int i = ty; i < 32; i += 8)
        tile[i][tx] = w[(size_t)(kb + i) * NDIM + nb + tx];
    __syncthreads();
    #pragma unroll
    for (int i = ty; i < 32; i += 8)
        wt[(size_t)(nb + i) * KDIM + kb + tx] = (__bf16)tile[tx][i];
}

__global__ __launch_bounds__(256) void zero_h_kernel(__bf16* __restrict__ h) {
    size_t i = (size_t)blockIdx.x * blockDim.x + threadIdx.x;
    h[i] = (__bf16)0.0f;
}

// ----------------------------- fused LSTM step -----------------------------
// Grid: 64 blocks; block owns d-slice [d0, d0+16) and all 4 gate strips.
// Block: 8 waves; wave w: gate g = w>>1, row-half mh = w&1; 2 accumulators.
// A panel [x_t|h] staged via double-buffered LDS chunks of 64x128 bf16.

__global__ __launch_bounds__(256) void lstm_step_kernel(
    const __bf16* __restrict__ xbf,    // [B, T, D] bf16
    const __bf16* __restrict__ wt,     // [4096, 2048] bf16
    const float*  __restrict__ bias,   // [4096]
    const int*    __restrict__ length, // [B]
    const __bf16* __restrict__ hin,    // [B, D] bf16  (h_{t-1})
    __bf16*       __restrict__ hout,   // [B, D] bf16  (h_t)
    float*        __restrict__ cst,    // [B, D] f32 cell state
    float*        __restrict__ hst,    // [B, D] f32 hidden state
    float*        __restrict__ out,    // ys [B,T,D] ++ c_fin [B,D] ++ h_fin [B,D]
    int t)
{
    __shared__ __bf16 atile[2][BATCH][APAD];  // ~34 KB double-buffered A panel
    __shared__ float  zs[4][BATCH][16];       // 16 KB gate exchange

    const int tid  = threadIdx.x;
    const int lane = tid & 31;
    const int wave = tid >> 5;
    const int g    = wave >> 1;           // gate 0..3
    const int mh   = wave & 1;            // row half
    const int d0   = blockIdx.x * 16;
    const int nl   = lane & 15;
    const int lk   = (lane >> 4) * 8;     // K sub-offset per WMMA 16-bit layout

    const int n = g * DIM + d0 + nl;      // Wt row for this lane's B column
    const __bf16* wrow = wt + (size_t)n * KDIM;

    // Loader mapping: thread -> 4 rows {rbase, +16, +32, +48}, quad qr in row.
    const int rbase = tid >> 4;           // 0..15
    const int qr    = tid & 15;           // 16B quad within 128-col chunk
    const __bf16* xsrc[4];
    const __bf16* hsrc[4];
    #pragma unroll
    for (int i = 0; i < 4; ++i) {
        const int r = rbase + 16 * i;
        xsrc[i] = xbf + ((size_t)r * TSTEPS + t) * DIM + qr * 8;
        hsrc[i] = hin + (size_t)r * DIM + qr * 8;
    }

    const int r0 = mh * 32 + nl;          // A rows of this wave's two M tiles
    const int r1 = r0 + 16;

    v8f acc0 = {};
    v8f acc1 = {};

    // Prologue: stage chunk 0 (x part, k offset 0).
    uint4 gld[4];
    #pragma unroll
    for (int i = 0; i < 4; ++i) gld[i] = *(const uint4*)(xsrc[i]);
    #pragma unroll
    for (int i = 0; i < 4; ++i)
        *(uint4*)&atile[0][rbase + 16 * i][qr * 8] = gld[i];
    __syncthreads();

    const int NCHUNK = KDIM / 128;        // 16 chunks of 4 k-iters
    for (int kc = 0; kc < NCHUNK; ++kc) {
        const int cur   = kc & 1;
        const int kbase = kc * 128;

        // (1) B chunk first (L2-resident weights) so its loadcnt retires first.
        Frag16 fbs[4];
        #pragma unroll
        for (int j = 0; j < 4; ++j) {
            fbs[j].q[0] = *(const uint4*)(wrow + kbase + j * 32 + lk);
            fbs[j].q[1] = *(const uint4*)(wrow + kbase + j * 32 + 16 + lk);
        }

        // (2) Issue next A chunk's global loads; no wait until the ds_store.
        if (kc + 1 < NCHUNK) {
            const int kn = kc + 1;
            #pragma unroll
            for (int i = 0; i < 4; ++i) {
                const __bf16* src = (kn < 8) ? (xsrc[i] + kn * 128)
                                             : (hsrc[i] + (kn - 8) * 128);
                gld[i] = *(const uint4*)src;
            }
        }

        // (3) 4 k-iters: A from LDS (ds_load_b128), 2 WMMA each.
        #pragma unroll
        for (int j = 0; j < 4; ++j) {
            const int kk = j * 32;
            Frag16 fa0, fa1;
            fa0.q[0] = *(const uint4*)&atile[cur][r0][kk + lk];
            fa0.q[1] = *(const uint4*)&atile[cur][r0][kk + 16 + lk];
            fa1.q[0] = *(const uint4*)&atile[cur][r1][kk + lk];
            fa1.q[1] = *(const uint4*)&atile[cur][r1][kk + 16 + lk];
            acc0 = __builtin_amdgcn_wmma_f32_16x16x32_bf16(false, fa0.v, false, fbs[j].v,
                                                           (short)0, acc0, false, false);
            acc1 = __builtin_amdgcn_wmma_f32_16x16x32_bf16(false, fa1.v, false, fbs[j].v,
                                                           (short)0, acc1, false, false);
        }

        // (4) Commit the prefetched chunk (loadcnt wait lands here, after WMMAs).
        if (kc + 1 < NCHUNK) {
            #pragma unroll
            for (int i = 0; i < 4; ++i)
                *(uint4*)&atile[1 - cur][rbase + 16 * i][qr * 8] = gld[i];
        }
        __syncthreads();
    }

    // C/D layout: vgpr e, lanes 0-15 -> M=e, lanes 16-31 -> M=8+e; N = lane&15
    const int mo = (lane >> 4) * 8;
    #pragma unroll
    for (int e = 0; e < 8; ++e) {
        zs[g][mh * 32 + mo + e][nl]      = acc0[e];
        zs[g][mh * 32 + 16 + mo + e][nl] = acc1[e];
    }
    __syncthreads();

    // Gates + state update for this block's (64 rows x 16 cols) slice.
    for (int idx = tid; idx < BATCH * 16; idx += 256) {
        const int b = idx >> 4;
        const int c = idx & 15;
        const int d = d0 + c;
        const size_t sidx = (size_t)b * DIM + d;

        float zi = zs[0][b][c] + bias[d];
        float zj = zs[1][b][c] + bias[DIM + d];
        float zf = zs[2][b][c] + bias[2 * DIM + d];
        float zo = zs[3][b][c] + bias[3 * DIM + d];

        float c_old = (t == 0) ? 0.0f : cst[sidx];
        float h_old = (t == 0) ? 0.0f : hst[sidx];

        float gi = 1.0f / (1.0f + expf(-zi));
        float gf = 1.0f / (1.0f + expf(-zf));
        float go = 1.0f / (1.0f + expf(-zo));
        float gj = tanhf(zj);

        float c_new = gf * c_old + gi * gj;
        float h_new = go * tanhf(c_new);

        const bool valid = t < length[b];
        float c2 = valid ? c_new : c_old;
        float h2 = valid ? h_new : h_old;

        out[((size_t)b * TSTEPS + t) * DIM + d] = valid ? h_new : 0.0f;
        cst[sidx] = c2;
        hst[sidx] = h2;
        hout[sidx] = (__bf16)h2;

        if (t == TSTEPS - 1) {
            const size_t ys = (size_t)BATCH * TSTEPS * DIM;
            out[ys + sidx] = c2;
            out[ys + (size_t)BATCH * DIM + sidx] = h2;
        }
    }
}

// ------------------------------- launcher ----------------------------------

extern "C" void kernel_launch(void* const* d_in, const int* in_sizes, int n_in,
                              void* d_out, int out_size, void* d_ws, size_t ws_size,
                              hipStream_t stream) {
    const float* x      = (const float*)d_in[0];   // [64,512,1024]
    const float* kern   = (const float*)d_in[1];   // [2048,4096]
    const float* bias   = (const float*)d_in[2];   // [4096]
    const int*   length = (const int*)d_in[3];     // [64]
    float* out = (float*)d_out;

    char* ws = (char*)d_ws;
    const size_t XBF_BYTES = (size_t)BATCH * TSTEPS * DIM * 2;  // 64 MB
    const size_t WT_BYTES  = (size_t)NDIM * KDIM * 2;           // 16 MB
    const size_t H_BYTES   = (size_t)BATCH * DIM * 2;           // 128 KB

    __bf16* xbf = (__bf16*)ws;
    __bf16* wt  = (__bf16*)(ws + XBF_BYTES);
    __bf16* h0  = (__bf16*)(ws + XBF_BYTES + WT_BYTES);
    __bf16* h1  = (__bf16*)(ws + XBF_BYTES + WT_BYTES + H_BYTES);
    float*  cst = (float*)(ws + XBF_BYTES + WT_BYTES + 2 * H_BYTES);
    float*  hst = cst + (size_t)BATCH * DIM;

    // One-time prep (re-run every call; deterministic).
    const size_t xquads = (size_t)BATCH * TSTEPS * DIM / 4;
    convert_x_kernel<<<(int)(xquads / 256), 256, 0, stream>>>((const float4*)x, xbf);
    dim3 tgrid(KDIM / 32, NDIM / 32);
    transpose_w_kernel<<<tgrid, 256, 0, stream>>>(kern, wt);
    zero_h_kernel<<<(BATCH * DIM) / 256, 256, 0, stream>>>(h0);

    // Sequential recurrence: one fused kernel per timestep, h double-buffered.
    for (int t = 0; t < TSTEPS; ++t) {
        __bf16* hin  = (t & 1) ? h1 : h0;
        __bf16* hout = (t & 1) ? h0 : h1;
        lstm_step_kernel<<<DIM / 16, 256, 0, stream>>>(
            xbf, wt, bias, length, hin, hout, cst, hst, out, t);
    }
}
